// FusedMoEGMM_26637387170051
// MI455X (gfx1250) — matrix operand order repaired and checked by reading, not compiled, not run on hardware
//
#include <hip/hip_runtime.h>
#include <hip/hip_bf16.h>

// CDNA5 / gfx1250 fused MoE: two-stage grouped GEMM with SwiGLU.
//   K1: g = silu(X@W13a^T) * (X@W13b^T)   (bf16 g -> workspace)
//   K2: out = g @ W2^T
// bf16 WMMA (v_wmma_f32_16x16x32_bf16), f32 accumulation, BM=64 weight reuse.
// Uniform fast-path epilogues: ragged (exec-masked) code only on boundary tiles.

typedef __bf16 bf16_t;
typedef __attribute__((ext_vector_type(16))) __bf16 v16bf;
typedef __attribute__((ext_vector_type(8)))  float  v8f;

constexpr int E = 8;
constexpr int T = 2048;
constexpr int H = 2048;
constexpr int I = 1408;

constexpr int NWAVES    = 8;           // 256 threads, wave32
constexpr int XS_STRIDE = H + 8;       // bf16 elems per LDS row (bank-conflict pad)
constexpr int GS_STRIDE = I + 8;

constexpr int BM  = 64;                // token rows per block (both kernels)
constexpr int MT  = T / BM;            // 32 worst-case tiles per expert
constexpr int NS1 = 4;                 // n-split, kernel 1
constexpr int JPB = (I / 16) / NS1;    // 22 j-chunks per block
constexpr int NS2 = 4;                 // n-split, kernel 2
constexpr int NPB = (H / 16) / NS2;    // 32 n-chunks per block

__device__ __forceinline__ v8f v8f_zero() {
  v8f v;
#pragma unroll
  for (int i = 0; i < 8; ++i) v[i] = 0.0f;
  return v;
}

__device__ __forceinline__ v8f wmma_bf16(v16bf a, v16bf b, v8f c) {
  return __builtin_amdgcn_wmma_f32_16x16x32_bf16(false, a, false, b, (short)0, c,
                                                 false, false);
}

// A-fragment (16x32 bf16, row-major in LDS). Contiguous layout (matches B):
// lane<16 -> row=lane, K in {kb..kb+15}, kb=(lane>>4)*16.
__device__ __forceinline__ v16bf lds_a_frag(const bf16_t* base, int stride,
                                            int mrow0, int kk) {
  const int lane = threadIdx.x & 31;
  const int row  = mrow0 + (lane & 15);
  const int kb   = kk + ((lane >> 4) << 4);
  const bf16_t* p0 = base + row * stride + kb;
  union { uint4 u[2]; v16bf v; } r;
  r.u[0] = *reinterpret_cast<const uint4*>(p0);        // ds_load_b128
  r.u[1] = *reinterpret_cast<const uint4*>(p0 + 8);
  return r.v;
}

// B-fragment (32x16 bf16): B[k][n] = W[(n0+n)*ldw + kk + k], W fp32 row-major.
// lane<16 -> col=lane, K=0..15; lane>=16 -> K=16..31. 4x global_load_b128 + cvt.
__device__ __forceinline__ v16bf gmem_b_frag(const float* __restrict__ wrows,
                                             int ldw, int kk) {
  const int lane = threadIdx.x & 31;
  const float* p = wrows + (size_t)(lane & 15) * (size_t)ldw + kk + ((lane >> 4) << 4);
  const float4* q = reinterpret_cast<const float4*>(p);
  union { float4 f4[4]; float f[16]; } t;
  t.f4[0] = q[0]; t.f4[1] = q[1]; t.f4[2] = q[2]; t.f4[3] = q[3];
  v16bf r;
#pragma unroll
  for (int i = 0; i < 16; ++i) r[i] = (bf16_t)t.f[i];
  return r;
}

__device__ __forceinline__ float silu_mul(float a, float b) {
  return (a / (1.0f + __expf(-a))) * b;
}

// ---------------- Kernel 1: g = SwiGLU(X @ W13^T) ----------------
__global__ __launch_bounds__(NWAVES * 32, 1)
void moe_gemm1_swiglu(const float* __restrict__ x,
                      const int*   __restrict__ expert_tokens,
                      const float* __restrict__ w13,
                      bf16_t*      __restrict__ gws) {
  extern __shared__ bf16_t Xs[];                       // [BM][XS_STRIDE]

  const int e = blockIdx.x / MT;
  const int m = blockIdx.x % MT;
  const int start = (e == 0) ? 0 : expert_tokens[e - 1];
  const int end   = expert_tokens[e];
  const int row0  = start + m * BM;
  if (row0 >= end) return;                             // uniform per block
  const int valid = min(BM, end - row0);
  const bool full = (valid == BM);

  const float* w13e = w13 + (size_t)e * (size_t)(2 * I) * H;

  // Stage X tile -> LDS bf16 (fast path: no row predicate).
  for (int idx = threadIdx.x; idx < BM * (H / 4); idx += blockDim.x) {
    const int r  = idx / (H / 4);
    const int c4 = idx % (H / 4);
    float4 v = make_float4(0.f, 0.f, 0.f, 0.f);
    if (full || r < valid)
      v = *reinterpret_cast<const float4*>(x + (size_t)(row0 + r) * H + c4 * 4);
    union { bf16_t h[4]; uint2 u; } pk;
    pk.h[0] = (bf16_t)v.x; pk.h[1] = (bf16_t)v.y;
    pk.h[2] = (bf16_t)v.z; pk.h[3] = (bf16_t)v.w;
    *reinterpret_cast<uint2*>(&Xs[r * XS_STRIDE + c4 * 4]) = pk.u;
  }
  __syncthreads();

  const int wave = threadIdx.x >> 5;
  const int lane = threadIdx.x & 31;
  const int half = lane >> 4;
  const int ncol = lane & 15;
  const int j0   = blockIdx.y * JPB;

  for (int j = j0 + wave; j < j0 + JPB; j += NWAVES) {
    v8f accG[4], accU[4];
#pragma unroll
    for (int mt = 0; mt < 4; ++mt) { accG[mt] = v8f_zero(); accU[mt] = v8f_zero(); }

    const float* wa = w13e + (size_t)(j * 16) * H;        // gate rows
    const float* wb = w13e + (size_t)(I + j * 16) * H;    // up rows

    // Prefetch first line of next j-panel handled by this wave.
    if (j + NWAVES < j0 + JPB) {
      __builtin_prefetch(wa + (size_t)(NWAVES * 16) * H + (size_t)lane * H, 0, 0);
      __builtin_prefetch(wb + (size_t)(NWAVES * 16) * H + (size_t)lane * H, 0, 0);
    }

    for (int kk = 0; kk < H; kk += 32) {
      v16bf Ba = gmem_b_frag(wa, H, kk);
      v16bf Bb = gmem_b_frag(wb, H, kk);
#pragma unroll
      for (int mt = 0; mt < 4; ++mt) {
        v16bf A = lds_a_frag(Xs, XS_STRIDE, mt * 16, kk);
        accG[mt] = wmma_bf16(A, Ba, accG[mt]);
        accU[mt] = wmma_bf16(A, Bb, accU[mt]);
      }
    }

    bf16_t* gbase = gws + (size_t)row0 * I + j * 16 + ncol;
    if (full) {
#pragma unroll
      for (int mt = 0; mt < 4; ++mt)
#pragma unroll
        for (int r = 0; r < 8; ++r) {
          const int rr = mt * 16 + half * 8 + r;
          gbase[(size_t)rr * I] = (bf16_t)silu_mul(accG[mt][r], accU[mt][r]);
        }
    } else {
#pragma unroll
      for (int mt = 0; mt < 4; ++mt)
#pragma unroll
        for (int r = 0; r < 8; ++r) {
          const int rr = mt * 16 + half * 8 + r;
          if (rr < valid)
            gbase[(size_t)rr * I] = (bf16_t)silu_mul(accG[mt][r], accU[mt][r]);
        }
    }
  }
}

// ---------------- Kernel 2: out = g @ W2^T ----------------
__global__ __launch_bounds__(NWAVES * 32, 1)
void moe_gemm2(const bf16_t* __restrict__ gws,
               const int*    __restrict__ expert_tokens,
               const float*  __restrict__ w2,
               float*        __restrict__ out) {
  extern __shared__ bf16_t Gs[];                       // [BM][GS_STRIDE]

  const int e = blockIdx.x / MT;
  const int m = blockIdx.x % MT;
  const int start = (e == 0) ? 0 : expert_tokens[e - 1];
  const int end   = expert_tokens[e];
  const int row0  = start + m * BM;
  if (row0 >= end) return;
  const int valid = min(BM, end - row0);
  const bool full = (valid == BM);

  const float* w2e = w2 + (size_t)e * (size_t)H * I;

  // Stage g tile -> LDS (fast path: no row predicate). 16B = 8 bf16 per move.
  for (int idx = threadIdx.x; idx < BM * (I / 8); idx += blockDim.x) {
    const int r = idx / (I / 8);
    const int c = idx % (I / 8);
    uint4 v = make_uint4(0u, 0u, 0u, 0u);
    if (full || r < valid)
      v = *reinterpret_cast<const uint4*>(gws + (size_t)(row0 + r) * I + c * 8);
    *reinterpret_cast<uint4*>(&Gs[r * GS_STRIDE + c * 8]) = v;
  }
  __syncthreads();

  const int wave = threadIdx.x >> 5;
  const int lane = threadIdx.x & 31;
  const int half = lane >> 4;
  const int ncol = lane & 15;
  const int n0   = blockIdx.y * NPB;

  for (int n = n0 + wave; n < n0 + NPB; n += NWAVES) {
    v8f c[4];
#pragma unroll
    for (int mt = 0; mt < 4; ++mt) c[mt] = v8f_zero();

    const float* wn = w2e + (size_t)(n * 16) * I;
    if (n + NWAVES < n0 + NPB)
      __builtin_prefetch(wn + (size_t)(NWAVES * 16) * I + (size_t)lane * I, 0, 0);

    for (int kk = 0; kk < I; kk += 32) {
      v16bf B = gmem_b_frag(wn, I, kk);
#pragma unroll
      for (int mt = 0; mt < 4; ++mt) {
        v16bf A = lds_a_frag(Gs, GS_STRIDE, mt * 16, kk);
        c[mt] = wmma_bf16(A, B, c[mt]);
      }
    }

    float* obase = out + (size_t)row0 * H + n * 16 + ncol;
    if (full) {
#pragma unroll
      for (int mt = 0; mt < 4; ++mt)
#pragma unroll
        for (int r = 0; r < 8; ++r) {
          const int rr = mt * 16 + half * 8 + r;
          obase[(size_t)rr * H] = c[mt][r];
        }
    } else {
#pragma unroll
      for (int mt = 0; mt < 4; ++mt)
#pragma unroll
        for (int r = 0; r < 8; ++r) {
          const int rr = mt * 16 + half * 8 + r;
          if (rr < valid) obase[(size_t)rr * H] = c[mt][r];
        }
    }
  }
}

extern "C" void kernel_launch(void* const* d_in, const int* in_sizes, int n_in,
                              void* d_out, int out_size, void* d_ws, size_t ws_size,
                              hipStream_t stream) {
  const float* x   = (const float*)d_in[0];
  const int*   et  = (const int*)d_in[1];
  const float* w13 = (const float*)d_in[2];
  const float* w2  = (const float*)d_in[3];
  float*       out = (float*)d_out;
  bf16_t*      gws = (bf16_t*)d_ws;          // T*I bf16 = 5.77 MB
  (void)in_sizes; (void)n_in; (void)out_size; (void)ws_size;

  const dim3 block(NWAVES * 32);
  const size_t smem1 = (size_t)BM * XS_STRIDE * sizeof(bf16_t);   // ~263 KB
  const size_t smem2 = (size_t)BM * GS_STRIDE * sizeof(bf16_t);   // ~181 KB

  hipLaunchKernelGGL(moe_gemm1_swiglu, dim3(E * MT, NS1), block, smem1, stream,
                     x, et, w13, gws);
  hipLaunchKernelGGL(moe_gemm2, dim3(E * MT, NS2), block, smem2, stream,
                     gws, et, w2, out);
}
